// MoEFFN_86131274154817
// MI455X (gfx1250) — compile-verified, matrix operand order checked by reading, not run on hardware
//
#include <hip/hip_runtime.h>
#include <cstdint>

// ---------------- problem constants ----------------
constexpr int Bc = 4, Tc = 2048, Dc = 512, Ec = 8, Hc = 2048;
constexpr int NTOK = Bc * Tc;          // 8192 tokens
constexpr int KC   = 64;               // H-chunk per iteration
constexpr int MT   = 32;               // tokens per workgroup tile
constexpr int NCHUNK = Hc / KC;        // 32
constexpr float MOE_COEFF = 0.01f;

// ---------------- WMMA types (CDNA5 / gfx1250, wave32) ----------------
typedef __attribute__((ext_vector_type(16))) __bf16 v16bf;
typedef __attribute__((ext_vector_type(8)))  float  v8f;

union BF16x16 { v16bf v; unsigned short s[16]; uint4 q[2]; };

// float -> bf16 bits, round-to-nearest-even (ignoring NaN edge cases)
__device__ __forceinline__ unsigned short f2bf(float f) {
    unsigned int u = __float_as_uint(f);
    u += 0x7FFFu + ((u >> 16) & 1u);
    return (unsigned short)(u >> 16);
}

// ---------------- TDM availability ----------------
#if defined(__has_builtin)
#if __has_builtin(__builtin_amdgcn_tensor_load_to_lds)
#define HAVE_TDM 1
#endif
#endif

#ifdef HAVE_TDM
typedef __attribute__((ext_vector_type(4))) unsigned int u32x4;
typedef __attribute__((ext_vector_type(8))) int i32x8;
typedef __attribute__((ext_vector_type(4))) int i32x4;

// One TDM descriptor: contiguous 64KB copy, data_size=8B, tile_dim0=8192 elems.
// Field packing per CDNA5 ISA (group0) / §8.4 (group1).
__device__ __forceinline__ void tdm_load_64kb(unsigned lds_addr, unsigned long long gaddr) {
    constexpr unsigned NE = 8192;                 // 64KB / 8B
    u32x4 g0;
    g0[0] = 1u;                                   // count=1, user mode
    g0[1] = lds_addr;                             // lds_addr[31:0]
    g0[2] = (unsigned)(gaddr & 0xFFFFFFFFull);    // global_addr[31:0]
    g0[3] = (unsigned)((gaddr >> 32) & 0x01FFFFFFull) | (2u << 30); // addr[56:32], type=2
    i32x8 g1;
    g1[0] = (int)(3u << 16);                      // workgroup_mask=0, data_size=3 (8B)
    g1[1] = (int)((NE & 0xFFFFu) << 16);          // tensor_dim0[15:0] @ bits[63:48]
    g1[2] = (int)((NE >> 16) | (1u << 16));       // tensor_dim0[31:16]; tensor_dim1=1
    g1[3] = (int)((NE & 0xFFFFu) << 16);          // tensor_dim1[31:16]=0; tile_dim0=8192
    g1[4] = 1;                                    // tile_dim1=1; tile_dim2=0
    g1[5] = (int)NE;                              // tensor_dim0_stride[31:0]
    g1[6] = 0; g1[7] = 0;
    i32x4 gz; gz[0] = 0; gz[1] = 0; gz[2] = 0; gz[3] = 0;
#if defined(__clang_major__) && (__clang_major__ >= 23)
    i32x8 gz8; gz8[0]=0; gz8[1]=0; gz8[2]=0; gz8[3]=0; gz8[4]=0; gz8[5]=0; gz8[6]=0; gz8[7]=0;
    __builtin_amdgcn_tensor_load_to_lds(g0, g1, gz, gz, gz8, 0);
#else
    __builtin_amdgcn_tensor_load_to_lds(g0, g1, gz, gz, 0);
#endif
}
#endif // HAVE_TDM

// =====================================================================
// Kernel 1: top-2 gate -> dense gate mask [NTOK,E] + aux-loss partials.
// =====================================================================
__global__ __launch_bounds__(256) void moe_gate_kernel(
    const float* __restrict__ x, const float* __restrict__ Wg,
    float* __restrict__ mask, float* __restrict__ partials)
{
    __shared__ float sWg[Dc * Ec];
    __shared__ float red[256 * 16];
    const int tid = threadIdx.x;
    for (int i = tid; i < Dc * Ec; i += 256) sWg[i] = Wg[i];
    __syncthreads();

    const int t = blockIdx.x * 256 + tid;
    float acc[Ec];
#pragma unroll
    for (int e = 0; e < Ec; ++e) acc[e] = 0.f;

    const float* xr = x + (size_t)t * Dc;
    for (int d = 0; d < Dc; ++d) {
        const float xv = xr[d];
        const float* wr = &sWg[d * Ec];
#pragma unroll
        for (int e = 0; e < Ec; ++e) acc[e] += xv * wr[e];
    }

    float mx = acc[0];
#pragma unroll
    for (int e = 1; e < Ec; ++e) mx = fmaxf(mx, acc[e]);
    float p[Ec]; float s = 0.f;
#pragma unroll
    for (int e = 0; e < Ec; ++e) { p[e] = __expf(acc[e] - mx); s += p[e]; }
    const float inv = 1.f / s;
#pragma unroll
    for (int e = 0; e < Ec; ++e) p[e] *= inv;

    int i0 = 0; float v0 = p[0];
#pragma unroll
    for (int e = 1; e < Ec; ++e) if (p[e] > v0) { v0 = p[e]; i0 = e; }
    int i1 = (i0 == 0) ? 1 : 0; float v1 = p[i1];
#pragma unroll
    for (int e = 0; e < Ec; ++e) if (e != i0 && p[e] > v1) { v1 = p[e]; i1 = e; }
    const float gd = 1.f / (v0 + v1);
    const float g0 = v0 * gd, g1 = v1 * gd;

#pragma unroll
    for (int e = 0; e < Ec; ++e)
        mask[(size_t)t * Ec + e] = (e == i0) ? g0 : ((e == i1) ? g1 : 0.f);

#pragma unroll
    for (int e = 0; e < Ec; ++e) {
        red[tid * 16 + e]     = p[e];
        red[tid * 16 + 8 + e] = (e == i0) ? 1.f : 0.f;
    }
    for (int st = 128; st > 0; st >>= 1) {
        __syncthreads();
        if (tid < st)
#pragma unroll
            for (int j = 0; j < 16; ++j)
                red[tid * 16 + j] += red[(tid + st) * 16 + j];
    }
    __syncthreads();
    if (tid < 16) partials[blockIdx.x * 16 + tid] = red[tid];
}

// =====================================================================
// Kernel 2: aux loss scalar -> d_out[NTOK*D]
// =====================================================================
__global__ void moe_aux_kernel(const float* __restrict__ partials,
                               float* __restrict__ out_aux)
{
    float imp[Ec], ld[Ec];
#pragma unroll
    for (int e = 0; e < Ec; ++e) { imp[e] = 0.f; ld[e] = 0.f; }
    for (int b = 0; b < NTOK / 256; ++b)
#pragma unroll
        for (int e = 0; e < Ec; ++e) {
            imp[e] += partials[b * 16 + e];
            ld[e]  += partials[b * 16 + 8 + e];
        }
    const float invN = 1.f / (float)NTOK;
    float s = 0.f;
#pragma unroll
    for (int e = 0; e < Ec; ++e) s += (imp[e] * invN) * (ld[e] * invN);
    *out_aux = (float)Ec * s * MOE_COEFF;
}

// =====================================================================
// Weight pre-transpose kernels (fp32 -> bf16, chunk-major layouts so each
// chunk is one contiguous 64KB block, TDM-copyable straight into LDS).
//   w1t[(e*32+c)*64 + hh][d]  = W1[e][d][c*64+hh]   (B of GEMM1, K=d contiguous)
//   w2t[(e*32+c)*512 + d][hh] = W2[e][c*64+hh][d]   (B of GEMM2, K=hh contiguous)
// LDS 64x(64+1) tiles keep both global sides coalesced.
// =====================================================================
__global__ __launch_bounds__(256) void transpose_w1_kernel(
    const float* __restrict__ W1, unsigned short* __restrict__ w1t)
{
    __shared__ unsigned short t[64 * 65];
    const int tid = threadIdx.x;
    const int blk = blockIdx.x;
    const int db = blk & 7, c = (blk >> 3) & 31, e = blk >> 8;
    const int d0 = db * 64, hc = c * 64;
    for (int i = tid; i < 64 * 64; i += 256) {
        const int r = i >> 6, h = i & 63;     // consecutive tid -> consecutive h (coalesced)
        const float v = W1[((size_t)e * Dc + d0 + r) * Hc + hc + h];
        t[h * 65 + r] = f2bf(v);
    }
    __syncthreads();
    for (int i = tid; i < 64 * 64; i += 256) {
        const int hh = i >> 6, dd = i & 63;   // consecutive tid -> consecutive d (coalesced)
        w1t[((size_t)(e * NCHUNK + c) * KC + hh) * Dc + d0 + dd] = t[hh * 65 + dd];
    }
}

__global__ __launch_bounds__(256) void transpose_w2_kernel(
    const float* __restrict__ W2, unsigned short* __restrict__ w2t)
{
    __shared__ unsigned short t[64 * 65];
    const int tid = threadIdx.x;
    const int blk = blockIdx.x;
    const int db = blk & 7, c = (blk >> 3) & 31, e = blk >> 8;
    const int d0 = db * 64, hc = c * 64;
    for (int i = tid; i < 64 * 64; i += 256) {
        const int hh = i >> 6, dd = i & 63;   // consecutive tid -> consecutive d (coalesced)
        const float v = W2[((size_t)e * Hc + hc + hh) * Dc + d0 + dd];
        t[dd * 65 + hh] = f2bf(v);
    }
    __syncthreads();
    for (int i = tid; i < 64 * 64; i += 256) {
        const int dd = i >> 6, hh = i & 63;   // consecutive tid -> consecutive hh (coalesced)
        w2t[((size_t)(e * NCHUNK + c) * Dc + d0 + dd) * KC + hh] = t[dd * 65 + hh];
    }
}

// =====================================================================
// WMMA core shared by both FFN kernels. STAGE=0: LDS already filled by TDM.
// STAGE=1: cooperative fp32->bf16 staging from the original weights.
// =====================================================================
template <int STAGE>
__device__ __forceinline__ void moe_ffn_body(
    const float* __restrict__ x,  const float* __restrict__ W1,
    const float* __restrict__ b1, const float* __restrict__ W2,
    const float* __restrict__ b2, const float* __restrict__ mask,
    const unsigned short* __restrict__ w1t, const unsigned short* __restrict__ w2t,
    float* __restrict__ y)
{
    extern __shared__ __align__(16) char smem[];
    unsigned short* w1s = (unsigned short*)smem;        // [KC][Dc] 64KB
    unsigned short* w2s = w1s + KC * Dc;                // [Dc][KC] 64KB
    unsigned short* hs  = w2s + Dc * KC;                // [MT][KC]  4KB

    const int tid  = threadIdx.x;
    const int lane = tid & 31;
    const int wave = tid >> 5;
    const int lh   = lane >> 4;      // half-lane: +8 K-offset per ISA fragment layout
    const int lm   = lane & 15;      // M (A) / N (B,C) index within a 16-tile

    const int tileBase = blockIdx.x * MT;
    const int mi1 = wave & 1, ni1 = wave >> 1;   // GEMM1: 2x4 tiling of h[32][64]
    const int mi2 = wave >> 2, nq  = wave & 3;   // GEMM2: 2x4 slabs of y[32][512]

    // preload x A-fragments (16 k-steps over K=512), ISA A-layout
    BF16x16 xa[16];
    {
        const int row = tileBase + mi1 * 16 + lm;
        const float2* xr = (const float2*)(x + (size_t)row * Dc);
#pragma unroll
        for (int k = 0; k < 16; ++k)
#pragma unroll
            for (int v = 0; v < 8; ++v) {
                const int kb = ((v < 4) ? (2 * v) : (16 + 2 * (v - 4))) + 8 * lh;
                const float2 w = xr[(k * 32 + kb) >> 1];
                xa[k].s[2 * v]     = f2bf(w.x);
                xa[k].s[2 * v + 1] = f2bf(w.y);
            }
    }

    v8f facc[8];
#pragma unroll
    for (int nt = 0; nt < 8; ++nt)
#pragma unroll
        for (int i = 0; i < 8; ++i) facc[nt][i] = 0.f;

    for (int e = 0; e < Ec; ++e) {
        float gm[8];
#pragma unroll
        for (int r = 0; r < 8; ++r)
            gm[r] = mask[(size_t)(tileBase + mi1 * 16 + 8 * lh + r) * Ec + e];

        const float* W1e = W1 + (size_t)e * Dc * Hc;
        const float* W2e = W2 + (size_t)e * Hc * Dc;

        for (int c = 0; c < NCHUNK; ++c) {
            const int hc = c * KC;

            if (STAGE) {
                // cooperative fp32->bf16 staging (fallback path)
                for (int i = tid; i < Dc * (KC / 4); i += 256) {
                    const int d = i >> 4, q = i & 15;
                    const float4 w = *(const float4*)(W1e + (size_t)d * Hc + hc + q * 4);
                    const int hh = q * 4;
                    w1s[(hh + 0) * Dc + d] = f2bf(w.x);
                    w1s[(hh + 1) * Dc + d] = f2bf(w.y);
                    w1s[(hh + 2) * Dc + d] = f2bf(w.z);
                    w1s[(hh + 3) * Dc + d] = f2bf(w.w);
                }
                for (int i = tid; i < KC * (Dc / 4); i += 256) {
                    const int hh = i >> 7, q = i & 127;
                    const int d0 = q * 4;
                    const float4 w = *(const float4*)(W2e + (size_t)(hc + hh) * Dc + d0);
                    w2s[(d0 + 0) * KC + hh] = f2bf(w.x);
                    w2s[(d0 + 1) * KC + hh] = f2bf(w.y);
                    w2s[(d0 + 2) * KC + hh] = f2bf(w.z);
                    w2s[(d0 + 3) * KC + hh] = f2bf(w.w);
                }
                if (c + 1 < NCHUNK && tid < 128) {
                    __builtin_prefetch(W1e + (size_t)(tid * 4) * Hc + hc + KC, 0, 1);
                    __builtin_prefetch(W2e + (size_t)(hc + KC) * Dc + tid * 4, 0, 1);
                }
            }
#ifdef HAVE_TDM
            else {
                // TDM path: wave 0 DMAs this chunk's pre-transposed bf16 blocks
                if (wave == 0) {
                    const size_t chunkOff = (size_t)(e * NCHUNK + c) * (KC * Dc * 2); // 64KB
                    tdm_load_64kb((unsigned)(uintptr_t)(void*)w1s,
                                  (unsigned long long)(uintptr_t)w1t + chunkOff);
                    tdm_load_64kb((unsigned)(uintptr_t)(void*)w2s,
                                  (unsigned long long)(uintptr_t)w2t + chunkOff);
                    __builtin_amdgcn_s_wait_tensorcnt(0);
                }
            }
#endif
            __syncthreads();

            // ---- GEMM1: this wave's 16x16 tile of h = x @ W1chunk (+b1), K=512
            v8f c1;
            {
                const float bv = b1[e * Hc + hc + ni1 * 16 + lm];
#pragma unroll
                for (int i = 0; i < 8; ++i) c1[i] = bv;
            }
            {
                const unsigned short* bcol = w1s + (ni1 * 16 + lm) * Dc;
#pragma unroll
                for (int k = 0; k < 16; ++k) {
                    const int d0 = k * 32 + 8 * lh;
                    BF16x16 bf;
                    bf.q[0] = *(const uint4*)(bcol + d0);
                    bf.q[1] = *(const uint4*)(bcol + d0 + 16);
                    c1 = __builtin_amdgcn_wmma_f32_16x16x32_bf16(
                        false, xa[k].v, false, bf.v, (short)0, c1, false, false);
                }
            }
            // ReLU + gate folded into h, bf16 store to LDS (C layout: M = r + 8*lh)
            {
                const int hh = ni1 * 16 + lm;
#pragma unroll
                for (int r = 0; r < 8; ++r) {
                    const int m = mi1 * 16 + 8 * lh + r;
                    const float v = c1[r] > 0.f ? c1[r] : 0.f;
                    hs[m * KC + hh] = f2bf(v * gm[r]);
                }
            }
            __syncthreads();

            // ---- GEMM2: facc += h_chunk @ W2chunk (K=64 -> 2 k-steps)
#pragma unroll
            for (int k2 = 0; k2 < 2; ++k2) {
                BF16x16 af;
                const unsigned short* arow = hs + (mi2 * 16 + lm) * KC + k2 * 32 + 8 * lh;
                af.q[0] = *(const uint4*)(arow);
                af.q[1] = *(const uint4*)(arow + 16);
#pragma unroll
                for (int nt = 0; nt < 8; ++nt) {
                    const int n = (nq * 8 + nt) * 16 + lm;
                    const unsigned short* brow = w2s + n * KC + k2 * 32 + 8 * lh;
                    BF16x16 bf2;
                    bf2.q[0] = *(const uint4*)(brow);
                    bf2.q[1] = *(const uint4*)(brow + 16);
                    facc[nt] = __builtin_amdgcn_wmma_f32_16x16x32_bf16(
                        false, af.v, false, bf2.v, (short)0, facc[nt], false, false);
                }
            }
            __syncthreads();
        }
    }

    // ---- epilogue: y = facc + sum_e gate[t,e]*b2[e]
#pragma unroll
    for (int nt = 0; nt < 8; ++nt) {
        const int d = (nq * 8 + nt) * 16 + lm;
#pragma unroll
        for (int r = 0; r < 8; ++r) {
            const int token = tileBase + mi2 * 16 + 8 * lh + r;
            const float* mrow = mask + (size_t)token * Ec;
            float b2c = 0.f;
#pragma unroll
            for (int ee = 0; ee < Ec; ++ee) b2c += mrow[ee] * b2[ee * Dc + d];
            y[(size_t)token * Dc + d] = facc[nt][r] + b2c;
        }
    }
}

__global__ __launch_bounds__(256) void moe_ffn_kernel_stage(
    const float* __restrict__ x,  const float* __restrict__ W1,
    const float* __restrict__ b1, const float* __restrict__ W2,
    const float* __restrict__ b2, const float* __restrict__ mask,
    float* __restrict__ y)
{
    moe_ffn_body<1>(x, W1, b1, W2, b2, mask, nullptr, nullptr, y);
}

#ifdef HAVE_TDM
__global__ __launch_bounds__(256) void moe_ffn_kernel_tdm(
    const float* __restrict__ x,  const float* __restrict__ W1,
    const float* __restrict__ b1, const float* __restrict__ W2,
    const float* __restrict__ b2, const float* __restrict__ mask,
    const unsigned short* __restrict__ w1t, const unsigned short* __restrict__ w2t,
    float* __restrict__ y)
{
    moe_ffn_body<0>(x, W1, b1, W2, b2, mask, w1t, w2t, y);
}
#endif

// =====================================================================
// launcher
// =====================================================================
extern "C" void kernel_launch(void* const* d_in, const int* in_sizes, int n_in,
                              void* d_out, int out_size, void* d_ws, size_t ws_size,
                              hipStream_t stream)
{
    const float* x  = (const float*)d_in[0];
    const float* Wg = (const float*)d_in[1];
    const float* W1 = (const float*)d_in[2];
    const float* b1 = (const float*)d_in[3];
    const float* W2 = (const float*)d_in[4];
    const float* b2 = (const float*)d_in[5];

    float* y        = (float*)d_out;                 // [NTOK*D] then aux scalar
    float* mask     = (float*)d_ws;                  // [NTOK*E]
    float* partials = mask + (size_t)NTOK * Ec;      // [(NTOK/256)*16]

    moe_gate_kernel<<<NTOK / 256, 256, 0, stream>>>(x, Wg, mask, partials);
    moe_aux_kernel<<<1, 1, 0, stream>>>(partials, y + (size_t)NTOK * Dc);

    const size_t lds_bytes = (size_t)(KC * Dc + Dc * KC + MT * KC) * sizeof(unsigned short); // 132KB

    // bf16 transposed weight images in scratch (16MB each), 512KB-aligned region
    const size_t W1T_OFF = (size_t)512 * 1024;
    const size_t WSLAB   = (size_t)Ec * NCHUNK * KC * Dc * sizeof(unsigned short); // 16MB
    const size_t W2T_OFF = W1T_OFF + WSLAB;
    const size_t WS_NEED = W2T_OFF + WSLAB;

#ifdef HAVE_TDM
    if (ws_size >= WS_NEED) {
        unsigned short* w1t = (unsigned short*)((char*)d_ws + W1T_OFF);
        unsigned short* w2t = (unsigned short*)((char*)d_ws + W2T_OFF);
        transpose_w1_kernel<<<Ec * NCHUNK * (Dc / 64), 256, 0, stream>>>(W1, w1t);
        transpose_w2_kernel<<<Ec * NCHUNK * (Dc / 64), 256, 0, stream>>>(W2, w2t);
        moe_ffn_kernel_tdm<<<NTOK / MT, 256, lds_bytes, stream>>>(
            x, W1, b1, W2, b2, mask, w1t, w2t, y);
        return;
    }
#endif
    moe_ffn_kernel_stage<<<NTOK / MT, 256, lds_bytes, stream>>>(x, W1, b1, W2, b2, mask, y);
}
